// RelativeAttention_29145648070828
// MI455X (gfx1250) — matrix operand order repaired and checked by reading, not compiled, not run on hardware
//
#include <hip/hip_runtime.h>
#include <math.h>

// Problem constants (from reference)
static constexpr int kB  = 2;
static constexpr int kS  = 2048;
static constexpr int kD  = 1024;
static constexpr int kH  = 16;
static constexpr int kHD = 64;
static constexpr int kNR = 65;   // 2*MAX_DIST+1
static constexpr int kMAXD = 32;

typedef __attribute__((ext_vector_type(16))) __bf16 v16bf;
typedef __attribute__((ext_vector_type(8)))  __bf16 v8bf;
typedef __attribute__((ext_vector_type(8)))  float  v8f;
typedef __attribute__((ext_vector_type(4)))  int    v4i;

typedef __attribute__((address_space(1))) v4i as1_v4i;   // global int4
typedef __attribute__((address_space(3))) v4i as3_v4i;   // LDS int4

// Async global->LDS 16-byte copy (ASYNCcnt-tracked DMA path on gfx1250);
// falls back to a plain vector copy if the builtin is unavailable.
static __device__ __forceinline__ void g2l_b128(const __bf16* gsrc, __bf16* ldst) {
#if __has_builtin(__builtin_amdgcn_global_load_async_to_lds_b128)
  __builtin_amdgcn_global_load_async_to_lds_b128(
      (as1_v4i*)(unsigned long long)(size_t)gsrc,
      (as3_v4i*)(unsigned int)(size_t)ldst, 0, 0);
#else
  *(v8bf*)ldst = *(const v8bf*)gsrc;
#endif
}
static __device__ __forceinline__ void wait_lds_arrivals() {
#if __has_builtin(__builtin_amdgcn_global_load_async_to_lds_b128)
#if __has_builtin(__builtin_amdgcn_s_wait_asynccnt)
  __builtin_amdgcn_s_wait_asynccnt(0);
#else
  asm volatile("s_wait_asynccnt 0" ::: "memory");
#endif
#endif
}

union FragU { v16bf v; v8bf h[2]; };

// A-matrix 16x32 bf16 fragment (row m = lane%16):
// lanes 0-15 hold K=[0..7],[16..23]; lanes 16-31 hold K=[8..15],[24..31]
static __device__ __forceinline__ v16bf frag_a(const __bf16* rowk, int lane) {
  const int o = (lane >> 4) * 8;
  FragU f;
  f.h[0] = *(const v8bf*)(rowk + o);
  f.h[1] = *(const v8bf*)(rowk + o + 16);
  return f.v;
}
// B-matrix 32x16 bf16 fragment (col n = lane%16):
// lanes 0-15 hold K=[0..7],[16..23]; lanes 16-31 hold K=[8..15],[24..31]
static __device__ __forceinline__ v16bf frag_b(const __bf16* colk, int lane) {
  const int o = (lane >> 4) * 8;
  FragU f;
  f.h[0] = *(const v8bf*)(colk + o);
  f.h[1] = *(const v8bf*)(colk + o + 16);
  return f.v;
}

// One-shot: f32 -> bf16 (pairs, so the compiler can use packed converts)
__global__ __launch_bounds__(256)
void cvt_bf16_kernel(const float* __restrict__ in, __bf16* __restrict__ out, int n2) {
  int i = (blockIdx.x * 256 + threadIdx.x) * 2;
  if (i < n2) {
    out[i]     = (__bf16)in[i];
    out[i + 1] = (__bf16)in[i + 1];
  }
}

// One-shot: W[K][N] f32 -> Wt[N][K] bf16 (LDS tile transpose, coalesced both sides)
__global__ __launch_bounds__(128)
void wtrans_kernel(const float* __restrict__ W, __bf16* __restrict__ Wt, int K, int N) {
  __shared__ __bf16 sT[64 * 72];
  const int tid = threadIdx.x;
  const int n0 = blockIdx.x * 64;
  const int k0 = blockIdx.y * 64;
#pragma unroll
  for (int i = 0; i < 32; ++i) {
    int idx = tid + i * 128;
    int k = idx >> 6, n = idx & 63;
    sT[n * 72 + k] = (__bf16)W[(size_t)(k0 + k) * N + n0 + n];
  }
  __syncthreads();
#pragma unroll
  for (int i = 0; i < 32; ++i) {
    int idx = tid + i * 128;
    int n = idx >> 6, k = idx & 63;
    Wt[(size_t)(n0 + n) * K + k0 + k] = sT[n * 72 + k];
  }
}

// C = alpha * (Xb @ Wt^T + bias). Xb:[M,K] bf16, Wt:[N,K] bf16 (pre-transposed).
// 64x64 tile / 128-thread block; staging is pure async b128 copies.
template <typename TO>
__global__ __launch_bounds__(128)
void gemm_bias_kernel(const __bf16* __restrict__ Xb, const __bf16* __restrict__ Wt,
                      const float* __restrict__ bias, TO* __restrict__ Cout,
                      int N, int K, float alpha) {
  __shared__ __align__(32) __bf16 sA[64 * 64];   // [m][k]
  __shared__ __align__(32) __bf16 sB[64 * 64];   // [n][k]
  const int tid  = threadIdx.x;
  const int wave = tid >> 5;
  const int lane = tid & 31;
  const int n0 = blockIdx.x * 64;
  const int m0 = blockIdx.y * 64;

  v8f acc[4];
#pragma unroll
  for (int t = 0; t < 4; ++t)
#pragma unroll
    for (int i = 0; i < 8; ++i) acc[t][i] = 0.0f;

  for (int kb = 0; kb < K; kb += 64) {
#pragma unroll
    for (int i = 0; i < 4; ++i) {
      int idx = tid + i * 128;            // 512 chunks of 8 elems
      int r = idx >> 3, c = (idx & 7) * 8;
      g2l_b128(&Xb[(size_t)(m0 + r) * K + kb + c], &sA[r * 64 + c]);
      g2l_b128(&Wt[(size_t)(n0 + r) * K + kb + c], &sB[r * 64 + c]);
    }
    wait_lds_arrivals();
    __syncthreads();
#pragma unroll
    for (int ks = 0; ks < 2; ++ks) {
      v16bf a = frag_a(&sA[(wave * 16 + (lane & 15)) * 64 + ks * 32], lane);
#pragma unroll
      for (int t = 0; t < 4; ++t) {
        v16bf bb = frag_b(&sB[(t * 16 + (lane & 15)) * 64 + ks * 32], lane);
        acc[t] = __builtin_amdgcn_wmma_f32_16x16x32_bf16(
            false, a, false, bb, (short)0, acc[t], false, false);
      }
    }
    __syncthreads();
  }
#pragma unroll
  for (int t = 0; t < 4; ++t) {
#pragma unroll
    for (int r = 0; r < 8; ++r) {
      int m = m0 + wave * 16 + r + ((lane >> 4) << 3);
      int n = n0 + t * 16 + (lane & 15);
      Cout[(size_t)m * N + n] = (TO)(alpha * (acc[t][r] + bias[n]));
    }
  }
}

// Flash attention with clipped relative-position bias (bf16 Q/K/V, f32 softmax).
// One block = (b, h, 64-row q block). 4 waves; wave w owns q rows [16w,16w+16).
__global__ __launch_bounds__(128)
void rel_attn_kernel(const __bf16* __restrict__ Qb, const __bf16* __restrict__ Kb,
                     const __bf16* __restrict__ Vb, const float* __restrict__ rel_emb,
                     __bf16* __restrict__ ctx) {
  __shared__ __align__(32) __bf16 sQ[64 * 64];   // [q][d]
  __shared__ __align__(32) __bf16 sK[64 * 64];   // [k][d]
  __shared__ __align__(32) __bf16 sV[64 * 64];   // transposed: [d][k]
  __shared__ __align__(32) __bf16 sP[64 * 64];   // [q][k]
  __shared__ float sRel[kNR * kHD];
  __shared__ float sQrel[64 * 66];               // [q][rel], padded stride

  const int tid  = threadIdx.x;
  const int wave = tid >> 5;
  const int lane = tid & 31;
  const int qb = blockIdx.x & 31;                // S/64 = 32
  const int h  = (blockIdx.x >> 5) & 15;
  const int b  = blockIdx.x >> 9;
  const int q0 = qb * 64;

  for (int idx = tid; idx < kNR * kHD; idx += 128) sRel[idx] = rel_emb[idx];
#pragma unroll
  for (int i = 0; i < 4; ++i) {
    int idx = tid + i * 128;
    int r = idx >> 3, c = (idx & 7) * 8;
    g2l_b128(&Qb[((size_t)(b * kS + q0 + r)) * kD + h * kHD + c], &sQ[r * 64 + c]);
  }
  wait_lds_arrivals();
  __syncthreads();

  // qrel[q][rel] = (scaled) Q row dot rel_emb row  (65 distinct rel rows)
  for (int idx = tid; idx < 64 * kNR; idx += 128) {
    int r = idx / kNR, dd = idx % kNR;
    float s = 0.f;
#pragma unroll 8
    for (int d = 0; d < kHD; ++d) s += (float)sQ[r * 64 + d] * sRel[dd * kHD + d];
    sQrel[r * 66 + dd] = s;
  }
  __syncthreads();

  v8f accO[4];
  float rowM[8], rowL[8];
#pragma unroll
  for (int t = 0; t < 4; ++t)
#pragma unroll
    for (int i = 0; i < 8; ++i) accO[t][i] = 0.0f;
#pragma unroll
  for (int r = 0; r < 8; ++r) { rowM[r] = -3.0e38f; rowL[r] = 0.0f; }

  for (int kt = 0; kt < kS / 64; ++kt) {
    const int k0 = kt * 64;
    const int knext = (kt + 1 < kS / 64) ? (k0 + 64) : k0;
    __builtin_prefetch(&Kb[((size_t)(b * kS + knext)) * kD + h * kHD], 0, 1);
#pragma unroll
    for (int i = 0; i < 4; ++i) {
      int idx = tid + i * 128;
      int r = idx >> 3, c = (idx & 7) * 8;
      size_t g = ((size_t)(b * kS + k0 + r)) * kD + h * kHD + c;
      g2l_b128(&Kb[g], &sK[r * 64 + c]);       // raw async copy
      v8bf vv = *(const v8bf*)&Vb[g];          // V: load + transpose into LDS
#pragma unroll
      for (int j = 0; j < 8; ++j) sV[(c + j) * 64 + r] = vv[j];
    }
    wait_lds_arrivals();
    __syncthreads();

    // S-tile = Qs @ K^T  (B fragment = K rows, contiguous along hd)
    v8f accS[4];
#pragma unroll
    for (int t = 0; t < 4; ++t)
#pragma unroll
      for (int i = 0; i < 8; ++i) accS[t][i] = 0.0f;
#pragma unroll
    for (int ks = 0; ks < 2; ++ks) {
      v16bf a = frag_a(&sQ[(wave * 16 + (lane & 15)) * 64 + ks * 32], lane);
#pragma unroll
      for (int t = 0; t < 4; ++t) {
        v16bf bb = frag_b(&sK[(t * 16 + (lane & 15)) * 64 + ks * 32], lane);
        accS[t] = __builtin_amdgcn_wmma_f32_16x16x32_bf16(
            false, a, false, bb, (short)0, accS[t], false, false);
      }
    }

    // add clipped relative bias, online softmax (rows live in 16-lane halves)
    float sv[4][8], tmax[8];
#pragma unroll
    for (int r = 0; r < 8; ++r) tmax[r] = -3.0e38f;
#pragma unroll
    for (int t = 0; t < 4; ++t) {
#pragma unroll
      for (int r = 0; r < 8; ++r) {
        int m  = r + ((lane >> 4) << 3);
        int qi = q0 + wave * 16 + m;
        int ki = k0 + t * 16 + (lane & 15);
        int dd = qi - ki;
        dd = dd < -kMAXD ? -kMAXD : (dd > kMAXD ? kMAXD : dd);
        float x = accS[t][r] + sQrel[(wave * 16 + m) * 66 + (dd + kMAXD)];
        sv[t][r] = x;
        tmax[r] = fmaxf(tmax[r], x);
      }
    }
#pragma unroll
    for (int r = 0; r < 8; ++r) {
      float v = tmax[r];
      v = fmaxf(v, __shfl_xor(v, 1, 32));
      v = fmaxf(v, __shfl_xor(v, 2, 32));
      v = fmaxf(v, __shfl_xor(v, 4, 32));
      v = fmaxf(v, __shfl_xor(v, 8, 32));
      tmax[r] = v;
    }
    float corr[8], psum[8];
#pragma unroll
    for (int r = 0; r < 8; ++r) {
      float nm = fmaxf(rowM[r], tmax[r]);
      corr[r] = __expf(rowM[r] - nm);
      rowM[r] = nm;
      psum[r] = 0.0f;
    }
#pragma unroll
    for (int t = 0; t < 4; ++t) {
#pragma unroll
      for (int r = 0; r < 8; ++r) {
        int m = r + ((lane >> 4) << 3);
        float p = __expf(sv[t][r] - rowM[r]);
        psum[r] += p;
        sP[(wave * 16 + m) * 64 + t * 16 + (lane & 15)] = (__bf16)p;
      }
    }
#pragma unroll
    for (int r = 0; r < 8; ++r) {
      float v = psum[r];
      v += __shfl_xor(v, 1, 32);
      v += __shfl_xor(v, 2, 32);
      v += __shfl_xor(v, 4, 32);
      v += __shfl_xor(v, 8, 32);
      rowL[r] = rowL[r] * corr[r] + v;
    }
#pragma unroll
    for (int t = 0; t < 4; ++t)
#pragma unroll
      for (int r = 0; r < 8; ++r) accO[t][r] *= corr[r];
    __syncthreads();   // sP visible to fragment reads

    // O += P @ V  (B fragment = sV columns, contiguous along k)
#pragma unroll
    for (int ks = 0; ks < 2; ++ks) {
      v16bf a = frag_a(&sP[(wave * 16 + (lane & 15)) * 64 + ks * 32], lane);
#pragma unroll
      for (int t = 0; t < 4; ++t) {
        v16bf bb = frag_b(&sV[(t * 16 + (lane & 15)) * 64 + ks * 32], lane);
        accO[t] = __builtin_amdgcn_wmma_f32_16x16x32_bf16(
            false, a, false, bb, (short)0, accO[t], false, false);
      }
    }
    __syncthreads();   // done with sK/sV/sP before next stage
  }

#pragma unroll
  for (int t = 0; t < 4; ++t) {
#pragma unroll
    for (int r = 0; r < 8; ++r) {
      int m = r + ((lane >> 4) << 3);
      int qi = q0 + wave * 16 + m;
      ctx[((size_t)(b * kS + qi)) * kD + h * kHD + t * 16 + (lane & 15)] =
          (__bf16)(accO[t][r] / rowL[r]);
    }
  }
}

extern "C" void kernel_launch(void* const* d_in, const int* in_sizes, int n_in,
                              void* d_out, int out_size, void* d_ws, size_t ws_size,
                              hipStream_t stream) {
  (void)in_sizes; (void)n_in; (void)out_size; (void)ws_size;
  const float* x   = (const float*)d_in[0];
  const float* Wq  = (const float*)d_in[1];
  const float* bq  = (const float*)d_in[2];
  const float* Wk  = (const float*)d_in[3];
  const float* bk  = (const float*)d_in[4];
  const float* Wv  = (const float*)d_in[5];
  const float* bv  = (const float*)d_in[6];
  const float* Wo  = (const float*)d_in[7];
  const float* bo  = (const float*)d_in[8];
  const float* rel = (const float*)d_in[9];
  float* out = (float*)d_out;

  const size_t T = (size_t)kB * kS;          // 4096 rows
  const size_t WSZ = (size_t)kD * kD;        // 1M elements per weight
  __bf16* xb  = (__bf16*)d_ws;               // [T,D]
  __bf16* WqT = xb  + T * kD;                // [N,K] bf16, pre-transposed
  __bf16* WkT = WqT + WSZ;
  __bf16* WvT = WkT + WSZ;
  __bf16* WoT = WvT + WSZ;
  __bf16* Q   = WoT + WSZ;                   // bf16, pre-scaled by 1/sqrt(hd)
  __bf16* Kp  = Q  + T * kD;
  __bf16* Vp  = Kp + T * kD;
  __bf16* Cx  = Vp + T * kD;

  const float scale = 0.125f;                // hd^-0.5, hd=64
  dim3 bl(128);
  dim3 gg(kD / 64, (unsigned)(T / 64));      // (16, 64)
  dim3 gw(kD / 64, kD / 64);                 // (16, 16)
  const int nx = (int)(T * kD);

  hipLaunchKernelGGL(cvt_bf16_kernel, dim3(nx / 512), dim3(256), 0, stream, x, xb, nx);
  hipLaunchKernelGGL(wtrans_kernel, gw, bl, 0, stream, Wq, WqT, kD, kD);
  hipLaunchKernelGGL(wtrans_kernel, gw, bl, 0, stream, Wk, WkT, kD, kD);
  hipLaunchKernelGGL(wtrans_kernel, gw, bl, 0, stream, Wv, WvT, kD, kD);
  hipLaunchKernelGGL(wtrans_kernel, gw, bl, 0, stream, Wo, WoT, kD, kD);

  hipLaunchKernelGGL((gemm_bias_kernel<__bf16>), gg, bl, 0, stream, xb, WqT, bq, Q,  kD, kD, scale);
  hipLaunchKernelGGL((gemm_bias_kernel<__bf16>), gg, bl, 0, stream, xb, WkT, bk, Kp, kD, kD, 1.0f);
  hipLaunchKernelGGL((gemm_bias_kernel<__bf16>), gg, bl, 0, stream, xb, WvT, bv, Vp, kD, kD, 1.0f);
  hipLaunchKernelGGL(rel_attn_kernel, dim3(kB * kH * (kS / 64)), bl, 0, stream, Q, Kp, Vp, rel, Cx);
  hipLaunchKernelGGL((gemm_bias_kernel<float>), gg, bl, 0, stream, Cx, WoT, bo, out, kD, kD, 1.0f);
}
